// FocalResidualAttention_75110388073131
// MI455X (gfx1250) — matrix operand order repaired and checked by reading, not compiled, not run on hardware
//
#include <hip/hip_runtime.h>

#define Bsz  2
#define Nseq 2048
#define DIM  1024
#define NH   16
#define HD   64
#define SCALE 0.125f   // HD^-0.5

typedef __bf16 bf16_t;
typedef __bf16 v16bf __attribute__((ext_vector_type(16)));
typedef float  v8f   __attribute__((ext_vector_type(8)));
typedef unsigned int u32x4 __attribute__((ext_vector_type(4)));
typedef int          i32x4 __attribute__((ext_vector_type(4)));
typedef int          i32x8 __attribute__((ext_vector_type(8)));

union AF { v16bf v; uint4 q[2]; };

static __device__ __forceinline__ v8f v8f_zero() {
  v8f z;
#pragma unroll
  for (int i = 0; i < 8; ++i) z[i] = 0.0f;
  return z;
}

static __device__ __forceinline__ v8f wmma_bf16(v16bf a, v16bf b, v8f c) {
  return __builtin_amdgcn_wmma_f32_16x16x32_bf16(false, a, false, b, (short)0, c, false, false);
}

// ---------------------------------------------------------------------------
// TDM: 2-D tile load global -> LDS (CDNA5 tensor_load_to_lds, 6-arg builtin).
// data_size = 8B units. tile row = tile_d0 units; row stride = stride0 units.
// LDS rows padded: pad_interval code i -> 2^(i+1) DWORDs, pad_amount code a ->
// (a+1) DWORDs appended. Descriptor packing per CDNA5 ISA ch.8 (D# groups).
// ---------------------------------------------------------------------------
static __device__ __forceinline__ void tdm_load_2d(const void* gaddr, unsigned lds_addr,
                                                   unsigned tile_d0_units, unsigned tile_d1,
                                                   unsigned stride0_units,
                                                   unsigned pad_i, unsigned pad_a)
{
  unsigned long long ga = (unsigned long long)(size_t)gaddr;
  u32x4 g0;
  g0[0] = 1u;                                                // count=1, user mode
  g0[1] = lds_addr;                                          // lds_addr [63:32]
  g0[2] = (unsigned)(ga & 0xffffffffu);                      // global_addr lo
  g0[3] = (unsigned)((ga >> 32) & 0x01ffffffu) | (2u << 30); // global_addr hi | type=2
  i32x8 g1;
  g1[0] = (int)((3u << 16) | (1u << 20) | (pad_i << 22) | (pad_a << 25)); // 8B, pad_en
  g1[1] = (int)((tile_d0_units & 0xffffu) << 16);            // tensor_dim0 lo
  g1[2] = (int)(((tile_d0_units >> 16) & 0xffffu) | ((tile_d1 & 0xffffu) << 16));
  g1[3] = (int)(((tile_d1 >> 16) & 0xffffu) | ((tile_d0_units & 0xffffu) << 16)); // tile_dim0
  g1[4] = (int)(tile_d1 & 0xffffu);                          // tile_dim1 (tile_dim2=0)
  g1[5] = (int)stride0_units;                                // tensor_dim0_stride lo
  g1[6] = 0;                                                 // stride hi / dim1_stride lo
  g1[7] = 0;
  i32x4 gz4; gz4[0] = 0; gz4[1] = 0; gz4[2] = 0; gz4[3] = 0;
  i32x8 gz8;
#pragma unroll
  for (int i = 0; i < 8; ++i) gz8[i] = 0;
  __builtin_amdgcn_tensor_load_to_lds(g0, g1, gz4, gz4, gz8, 0);
}

static __device__ __forceinline__ unsigned lds_off(const void* p) {
  return (unsigned)(size_t)p;  // generic LDS pointer: low 32 bits = LDS offset
}

// ---------------------------------------------------------------------------
// fp32 -> bf16 convert, 8 elements / thread
// ---------------------------------------------------------------------------
__global__ void cvt_f32_bf16(const float* __restrict__ in, bf16_t* __restrict__ out, int n) {
  int i = (blockIdx.x * blockDim.x + threadIdx.x) * 8;
  if (i < n) {
    float4 a = *reinterpret_cast<const float4*>(in + i);
    float4 b = *reinterpret_cast<const float4*>(in + i + 4);
    union { bf16_t h[8]; uint4 u; } pk;
    pk.h[0] = (bf16_t)a.x; pk.h[1] = (bf16_t)a.y; pk.h[2] = (bf16_t)a.z; pk.h[3] = (bf16_t)a.w;
    pk.h[4] = (bf16_t)b.x; pk.h[5] = (bf16_t)b.y; pk.h[6] = (bf16_t)b.z; pk.h[7] = (bf16_t)b.w;
    *reinterpret_cast<uint4*>(out + i) = pk.u;
  }
}

// ---------------------------------------------------------------------------
// bf16 NT GEMM: C[M,Nn] = A[M,K] * W[Nn,K]^T + bias[Nn]
// TDM double-buffered tiles, 8 waves, tile 128x128, K-step 32.
// mode 0: f32 out; mode 1: QKV scatter (Q/K row-major, V transposed).
// ---------------------------------------------------------------------------
#define MT 128
#define NTL 128
#define KTS 32
#define LDP 40   // 32 elements + 8 pad (TDM: 16-DW interval, 4-DW pad)

__global__ __launch_bounds__(256)
void gemm_bf16_nt(const bf16_t* __restrict__ A, const bf16_t* __restrict__ Bw,
                  const float* __restrict__ bias, int M, int Nn, int K,
                  int mode, float* __restrict__ Cout,
                  bf16_t* __restrict__ Qo, bf16_t* __restrict__ Ko, bf16_t* __restrict__ Vto)
{
  __shared__ bf16_t As[2][MT * LDP];
  __shared__ bf16_t Bs[2][NTL * LDP];

  const int tid  = threadIdx.x;
  const int lane = tid & 31;
  const int wave = tid >> 5;      // 0..7
  const int wm   = wave >> 2;     // 0..1
  const int wn   = wave & 3;      // 0..3
  const int lm   = lane & 15;
  const int khi  = lane >> 4;
  const int m0   = blockIdx.y * MT;
  const int n0   = blockIdx.x * NTL;

  v8f acc[4][2];
#pragma unroll
  for (int i = 0; i < 4; ++i)
#pragma unroll
    for (int j = 0; j < 2; ++j) acc[i][j] = v8f_zero();

  // Prologue: TDM tile 0 into buffer 0 (wave0: A, wave1: B)
  if (wave == 0)
    tdm_load_2d(A + (size_t)m0 * K, lds_off(&As[0][0]), KTS / 4, MT, (unsigned)(K / 4), 3, 3);
  else if (wave == 1)
    tdm_load_2d(Bw + (size_t)n0 * K, lds_off(&Bs[0][0]), KTS / 4, NTL, (unsigned)(K / 4), 3, 3);
  if (wave <= 1) __builtin_amdgcn_s_wait_tensorcnt(0);
  __syncthreads();

  int cur = 0;
  for (int k0 = 0; k0 < K; k0 += KTS) {
    const int nxt = cur ^ 1;
    if (k0 + KTS < K) {  // pipeline: DMA next tile while computing this one
      if (wave == 0)
        tdm_load_2d(A + (size_t)m0 * K + k0 + KTS, lds_off(&As[nxt][0]),
                    KTS / 4, MT, (unsigned)(K / 4), 3, 3);
      else if (wave == 1)
        tdm_load_2d(Bw + (size_t)n0 * K + k0 + KTS, lds_off(&Bs[nxt][0]),
                    KTS / 4, NTL, (unsigned)(K / 4), 3, 3);
    }

    AF af[4], bfr[2];
#pragma unroll
    for (int ms = 0; ms < 4; ++ms) {
      const bf16_t* base = &As[cur][0] + (wm * 64 + ms * 16 + lm) * LDP + khi * 8;
      af[ms].q[0] = *reinterpret_cast<const uint4*>(base);
      af[ms].q[1] = *reinterpret_cast<const uint4*>(base + 16);
    }
#pragma unroll
    for (int ns = 0; ns < 2; ++ns) {
      const bf16_t* base = &Bs[cur][0] + (wn * 32 + ns * 16 + lm) * LDP + khi * 8;
      bfr[ns].q[0] = *reinterpret_cast<const uint4*>(base);
      bfr[ns].q[1] = *reinterpret_cast<const uint4*>(base + 16);
    }
#pragma unroll
    for (int ms = 0; ms < 4; ++ms)
#pragma unroll
      for (int ns = 0; ns < 2; ++ns)
        acc[ms][ns] = wmma_bf16(af[ms].v, bfr[ns].v, acc[ms][ns]);

    if (wave <= 1) __builtin_amdgcn_s_wait_tensorcnt(0);
    __syncthreads();
    cur = nxt;
  }

#pragma unroll
  for (int ms = 0; ms < 4; ++ms)
#pragma unroll
    for (int ns = 0; ns < 2; ++ns)
#pragma unroll
      for (int r = 0; r < 8; ++r) {
        const int row = m0 + wm * 64 + ms * 16 + r + 8 * khi;
        const int col = n0 + wn * 32 + ns * 16 + lm;
        const float v = acc[ms][ns][r] + bias[col];
        if (mode == 0) {
          Cout[(size_t)row * Nn + col] = v;
        } else {
          const int b = row / Nseq, n = row % Nseq;
          const int s = col / DIM,  rem = col % DIM;
          const int h = rem / HD,   d = rem % HD;
          const size_t bh = (size_t)(b * NH + h);
          const bf16_t bv = (bf16_t)v;
          if (s == 0)      Qo[(bh * Nseq + n) * HD + d] = bv;
          else if (s == 1) Ko[(bh * Nseq + n) * HD + d] = bv;
          else             Vto[(bh * HD + d) * Nseq + n] = bv;
        }
      }
}

// ---------------------------------------------------------------------------
// Focal attention, flash-style with two exponent tracks:
//   out = alpha*O / (alpha*Z + 1e-6*l^g);  O=sum e^{g(s-m)}v, Z=sum e^{g(s-m)},
//   l = sum e^{s-m}
// 4 waves, 64 queries/block, TDM double-buffered 64-key K/V tiles.
// ---------------------------------------------------------------------------
#define QT   64
#define KTL  64
#define LKP  72  // 64 elements + 8 pad (TDM: 32-DW interval, 4-DW pad)

__global__ __launch_bounds__(128)
void focal_attn(const bf16_t* __restrict__ Qg, const bf16_t* __restrict__ Kg,
                const bf16_t* __restrict__ Vtg, bf16_t* __restrict__ AO,
                const float* __restrict__ alpha_p, const float* __restrict__ gamma_p)
{
  __shared__ bf16_t Ks[2][KTL * LKP];     // [key][hd]
  __shared__ bf16_t Vs[2][KTL * LKP];     // [hd][key]
  __shared__ bf16_t Ps[4 * 16 * LKP];     // per-wave P tile [q][key]

  const int tid  = threadIdx.x;
  const int lane = tid & 31;
  const int wave = tid >> 5;     // 0..3
  const int lm   = lane & 15;
  const int khi  = lane >> 4;
  const int bh   = blockIdx.x;   // b*NH + h
  const int b    = bh / NH, h = bh % NH;
  const int q0   = blockIdx.y * QT + wave * 16;

  const float alpha = alpha_p[0];
  const float g     = gamma_p[0];

  const size_t qkBase = (size_t)bh * Nseq * HD;
  const size_t vtBase = (size_t)bh * HD * Nseq;

  // Preload Q fragments (2 K-steps over HD=64)
  AF qf[2];
#pragma unroll
  for (int ks = 0; ks < 2; ++ks) {
    const bf16_t* base = Qg + qkBase + (size_t)(q0 + lm) * HD + ks * 32 + khi * 8;
    qf[ks].q[0] = *reinterpret_cast<const uint4*>(base);
    qf[ks].q[1] = *reinterpret_cast<const uint4*>(base + 16);
  }

  v8f o[4];
#pragma unroll
  for (int t = 0; t < 4; ++t) o[t] = v8f_zero();
  float mrow[8], lrow[8], Zrow[8];
#pragma unroll
  for (int r = 0; r < 8; ++r) { mrow[r] = -3.0e38f; lrow[r] = 0.0f; Zrow[r] = 0.0f; }

  bf16_t* pw = Ps + wave * 16 * LKP;

  // Prologue TDM: K tile rows=keys (stride HD), V tile rows=hd (stride Nseq)
  if (wave == 0)
    tdm_load_2d(Kg + qkBase, lds_off(&Ks[0][0]), HD / 4, KTL, HD / 4, 4, 3);
  else if (wave == 1)
    tdm_load_2d(Vtg + vtBase, lds_off(&Vs[0][0]), KTL / 4, HD, Nseq / 4, 4, 3);
  if (wave <= 1) __builtin_amdgcn_s_wait_tensorcnt(0);
  __syncthreads();

  int cur = 0;
  for (int k0 = 0; k0 < Nseq; k0 += KTL) {
    const int nxt = cur ^ 1;
    if (k0 + KTL < Nseq) {
      if (wave == 0)
        tdm_load_2d(Kg + qkBase + (size_t)(k0 + KTL) * HD, lds_off(&Ks[nxt][0]),
                    HD / 4, KTL, HD / 4, 4, 3);
      else if (wave == 1)
        tdm_load_2d(Vtg + vtBase + k0 + KTL, lds_off(&Vs[nxt][0]),
                    KTL / 4, HD, Nseq / 4, 4, 3);
    }

    // S = (Q K^T) * SCALE : 4 key-subtiles of 16
    v8f s[4];
#pragma unroll
    for (int t = 0; t < 4; ++t) {
      AF bf0, bf1;
      const bf16_t* kb = &Ks[cur][0] + (t * 16 + lm) * LKP;
      bf0.q[0] = *reinterpret_cast<const uint4*>(kb + khi * 8);
      bf0.q[1] = *reinterpret_cast<const uint4*>(kb + khi * 8 + 16);
      bf1.q[0] = *reinterpret_cast<const uint4*>(kb + 32 + khi * 8);
      bf1.q[1] = *reinterpret_cast<const uint4*>(kb + 32 + khi * 8 + 16);
      v8f z = v8f_zero();
      z = wmma_bf16(qf[0].v, bf0.v, z);
      z = wmma_bf16(qf[1].v, bf1.v, z);
#pragma unroll
      for (int r = 0; r < 8; ++r) z[r] *= SCALE;
      s[t] = z;
    }

    // Online stats (lanes 0-15 -> rows r, lanes 16-31 -> rows r+8)
    float la[8], za[8];
#pragma unroll
    for (int r = 0; r < 8; ++r) {
      float v = s[0][r];
      v = fmaxf(v, s[1][r]); v = fmaxf(v, s[2][r]); v = fmaxf(v, s[3][r]);
#pragma unroll
      for (int msk = 1; msk < 16; msk <<= 1) v = fmaxf(v, __shfl_xor(v, msk, 32));
      const float mn = fmaxf(mrow[r], v);
      const float c1 = __expf(mrow[r] - mn);
      const float cg = __expf(g * (mrow[r] - mn));
      lrow[r] *= c1;  Zrow[r] *= cg;
#pragma unroll
      for (int t = 0; t < 4; ++t) o[t][r] *= cg;
      mrow[r] = mn;
      la[r] = 0.0f; za[r] = 0.0f;
    }

    // P = e^{g(s-m)} -> per-wave LDS (bf16), accumulate l/Z partials
#pragma unroll
    for (int t = 0; t < 4; ++t) {
#pragma unroll
      for (int r = 0; r < 8; ++r) {
        const float d0 = s[t][r] - mrow[r];
        const float p1 = __expf(d0);
        const float pf = __expf(g * d0);
        la[r] += p1; za[r] += pf;
        pw[(r + 8 * khi) * LKP + t * 16 + lm] = (bf16_t)pf;
      }
    }
#pragma unroll
    for (int r = 0; r < 8; ++r) {
      float lv = la[r], zv = za[r];
#pragma unroll
      for (int msk = 1; msk < 16; msk <<= 1) {
        lv += __shfl_xor(lv, msk, 32);
        zv += __shfl_xor(zv, msk, 32);
      }
      lrow[r] += lv; Zrow[r] += zv;
    }

    asm volatile("" ::: "memory");  // same-wave LDS is in-order; stop reordering

    // O += P(16x64) * V(64x64)
    AF pa[2];
#pragma unroll
    for (int ks = 0; ks < 2; ++ks) {
      const bf16_t* base = pw + lm * LKP + ks * 32 + khi * 8;
      pa[ks].q[0] = *reinterpret_cast<const uint4*>(base);
      pa[ks].q[1] = *reinterpret_cast<const uint4*>(base + 16);
    }
#pragma unroll
    for (int t = 0; t < 4; ++t) {
      AF vb0, vb1;
      const bf16_t* vbse = &Vs[cur][0] + (t * 16 + lm) * LKP;
      vb0.q[0] = *reinterpret_cast<const uint4*>(vbse + khi * 8);
      vb0.q[1] = *reinterpret_cast<const uint4*>(vbse + khi * 8 + 16);
      vb1.q[0] = *reinterpret_cast<const uint4*>(vbse + 32 + khi * 8);
      vb1.q[1] = *reinterpret_cast<const uint4*>(vbse + 32 + khi * 8 + 16);
      o[t] = wmma_bf16(pa[0].v, vb0.v, o[t]);
      o[t] = wmma_bf16(pa[1].v, vb1.v, o[t]);
    }

    if (wave <= 1) __builtin_amdgcn_s_wait_tensorcnt(0);
    __syncthreads();
    cur = nxt;
  }

  // out = alpha*O / (alpha*Z + 1e-6 * l^g)
  float denom[8];
#pragma unroll
  for (int r = 0; r < 8; ++r) {
    const float lg = __expf(g * __logf(lrow[r] + 1e-30f));
    denom[r] = alpha * Zrow[r] + 1e-6f * lg;
  }
#pragma unroll
  for (int t = 0; t < 4; ++t)
#pragma unroll
    for (int r = 0; r < 8; ++r) {
      const int qrow = q0 + r + 8 * khi;
      const int dcol = t * 16 + lm;
      const float v = alpha * o[t][r] / denom[r];
      AO[((size_t)(b * Nseq) + qrow) * DIM + h * HD + dcol] = (bf16_t)v;
    }
}

// ---------------------------------------------------------------------------
extern "C" void kernel_launch(void* const* d_in, const int* in_sizes, int n_in,
                              void* d_out, int out_size, void* d_ws, size_t ws_size,
                              hipStream_t stream)
{
  (void)in_sizes; (void)n_in; (void)out_size; (void)ws_size;
  const float* x     = (const float*)d_in[0];
  const float* Wqkv  = (const float*)d_in[1];
  const float* bqkv  = (const float*)d_in[2];
  const float* Wproj = (const float*)d_in[3];
  const float* bproj = (const float*)d_in[4];
  const float* alpha = (const float*)d_in[5];
  const float* gamma = (const float*)d_in[6];
  float* out = (float*)d_out;

  char* ws = (char*)d_ws;
  size_t off = 0;
  auto carve = [&](size_t bytes) {
    void* p = ws + off;
    off = (off + bytes + 255) & ~(size_t)255;
    return p;
  };
  const int MBN = Bsz * Nseq;                     // 4096
  bf16_t* Xbf   = (bf16_t*)carve((size_t)MBN * DIM * 2);
  bf16_t* Wqkvb = (bf16_t*)carve((size_t)3 * DIM * DIM * 2);
  bf16_t* Wprjb = (bf16_t*)carve((size_t)DIM * DIM * 2);
  bf16_t* Qbf   = (bf16_t*)carve((size_t)MBN * DIM * 2);
  bf16_t* Kbf   = (bf16_t*)carve((size_t)MBN * DIM * 2);
  bf16_t* Vtbf  = (bf16_t*)carve((size_t)MBN * DIM * 2);
  bf16_t* AObf  = (bf16_t*)carve((size_t)MBN * DIM * 2);

  const int nx = MBN * DIM, nwq = 3 * DIM * DIM, nwp = DIM * DIM;
  cvt_f32_bf16<<<(nx / 8 + 255) / 256, 256, 0, stream>>>(x,     Xbf,   nx);
  cvt_f32_bf16<<<(nwq / 8 + 255) / 256, 256, 0, stream>>>(Wqkv,  Wqkvb, nwq);
  cvt_f32_bf16<<<(nwp / 8 + 255) / 256, 256, 0, stream>>>(Wproj, Wprjb, nwp);

  // QKV projection: [4096, 3072] = X @ Wqkv^T + bqkv, scattered to Q/K/Vt
  dim3 gq(3 * DIM / NTL, MBN / MT);
  gemm_bf16_nt<<<gq, 256, 0, stream>>>(Xbf, Wqkvb, bqkv, MBN, 3 * DIM, DIM,
                                       1, nullptr, Qbf, Kbf, Vtbf);

  // Focal attention
  dim3 ga(Bsz * NH, Nseq / QT);
  focal_attn<<<ga, 128, 0, stream>>>(Qbf, Kbf, Vtbf, AObf, alpha, gamma);

  // Output projection: [4096, 1024] = AO @ Wproj^T + bproj (f32 -> d_out)
  dim3 gp(DIM / NTL, MBN / MT);
  gemm_bf16_nt<<<gp, 256, 0, stream>>>(AObf, Wprjb, bproj, MBN, DIM, DIM,
                                       0, out, nullptr, nullptr, nullptr);
}
